// Dit_block_49520972923384
// MI455X (gfx1250) — compile-verified, matrix operand order checked by reading, not run on hardware
//
#include <hip/hip_runtime.h>
#include <hip/hip_bf16.h>
#include <stdint.h>

typedef __attribute__((ext_vector_type(16))) __bf16 v16bf;
typedef __attribute__((ext_vector_type(8)))  float  v8f;
typedef unsigned short u16;

#define Bb 8
#define Ll 1024
#define Cc 1152
#define Hh 16
#define Dd 72
#define Dp 96
#define C4 4608
#define C6 6912
#define Mr 8192   // B*L

__device__ __forceinline__ u16 f2bf(float f) {
    unsigned u = __float_as_uint(f);
    u += 0x7fffu + ((u >> 16) & 1u);
    return (u16)(u >> 16);
}

union Frag { v16bf v; uint4 q[2]; u16 h[16]; };
union Acc  { v8f v; float f[8]; };

__device__ __forceinline__ v8f wmma_bf16(v16bf a, v16bf b, v8f c) {
    return __builtin_amdgcn_wmma_f32_16x16x32_bf16(false, a, false, b, (short)0, c, false, false);
}

// ---------------------------------------------------------------- fp32 [K][N] -> bf16 [N][K] (tiled transpose)
__global__ __launch_bounds__(256)
void transpose_f2bf_kernel(const float* __restrict__ src, u16* __restrict__ dst,
                           int K, int N) {
    __shared__ float tile[32][33];
    const int k0 = blockIdx.x * 32;
    const int n0 = blockIdx.y * 32;
    const int tid = threadIdx.x;
    const int tr = tid >> 5, tc = tid & 31;
    #pragma unroll
    for (int i = 0; i < 4; ++i)
        tile[tr + i * 8][tc] = src[(size_t)(k0 + tr + i * 8) * N + n0 + tc];
    __syncthreads();
    #pragma unroll
    for (int i = 0; i < 4; ++i)
        dst[(size_t)(n0 + tr + i * 8) * K + k0 + tc] = f2bf(tile[tc][tr + i * 8]);
}

// ---------------------------------------------------------------- adaLN: silu(cond) @ ada_w + ada_b -> [8, 6C]
__global__ __launch_bounds__(256)
void ada_kernel(const float* __restrict__ condition, const float* __restrict__ adaw,
                const float* __restrict__ adab, float* __restrict__ out) {
    __shared__ float sact[Bb * Cc];
    int tid = threadIdx.x;
    for (int idx = tid; idx < Bb * Cc; idx += 256) {
        int b = idx / Cc, k = idx - b * Cc;
        float x = condition[b * Cc + k];
        sact[idx] = x / (1.f + __expf(-x));
    }
    __syncthreads();
    int c = blockIdx.x * 256 + tid;
    float acc[Bb];
    #pragma unroll
    for (int b = 0; b < Bb; ++b) acc[b] = 0.f;
    for (int k = 0; k < Cc; ++k) {
        float w = adaw[(size_t)k * C6 + c];
        #pragma unroll
        for (int b = 0; b < Bb; ++b) acc[b] = fmaf(sact[b * Cc + k], w, acc[b]);
    }
    float bias = adab[c];
    #pragma unroll
    for (int b = 0; b < Bb; ++b) out[b * C6 + c] = acc[b] + bias;
}

// ---------------------------------------------------------------- LN1 + modulate -> bf16
__global__ __launch_bounds__(256)
void ln1_mod_kernel(const float* __restrict__ x, const float* __restrict__ cond,
                    const float* __restrict__ lnw, const float* __restrict__ lnb,
                    u16* __restrict__ hbf) {
    __shared__ float red[256];
    int row = blockIdx.x, b = row >> 10, tid = threadIdx.x;
    const float* xr = x + (size_t)row * Cc;
    float v[5], s = 0.f, s2 = 0.f;
    #pragma unroll
    for (int i = 0; i < 5; ++i) {
        int c = tid + i * 256;
        float t = (c < Cc) ? xr[c] : 0.f;
        v[i] = t; s += t; s2 += t * t;
    }
    red[tid] = s; __syncthreads();
    for (int st = 128; st > 0; st >>= 1) { if (tid < st) red[tid] += red[tid + st]; __syncthreads(); }
    float mu = red[0] * (1.f / Cc); __syncthreads();
    red[tid] = s2; __syncthreads();
    for (int st = 128; st > 0; st >>= 1) { if (tid < st) red[tid] += red[tid + st]; __syncthreads(); }
    float var = red[0] * (1.f / Cc) - mu * mu;
    float rs = rsqrtf(var + 1e-6f);
    const float* cb = cond + b * C6;
    #pragma unroll
    for (int i = 0; i < 5; ++i) {
        int c = tid + i * 256;
        if (c < Cc) {
            float y = (v[i] - mu) * rs * lnw[c] + lnb[c];
            y = y * (1.f + cb[4 * Cc + c]) + cb[2 * Cc + c];   // gamma1, beta1
            hbf[(size_t)row * Cc + c] = f2bf(y);
        }
    }
}

// ---------------------------------------------------------------- residual (reshape-bug gather) + LN2 + modulate
__global__ __launch_bounds__(256)
void resid_ln2_kernel(const float* __restrict__ x, const float* __restrict__ attn,
                      const float* __restrict__ cond, const float* __restrict__ lnw,
                      const float* __restrict__ lnb, float* __restrict__ x1,
                      u16* __restrict__ h2) {
    __shared__ float red[256];
    int row = blockIdx.x, b = row >> 10, l = row & 1023, tid = threadIdx.x;
    const float* cb = cond + b * C6;
    float v[5], s = 0.f, s2 = 0.f;
    #pragma unroll
    for (int i = 0; i < 5; ++i) {
        int c = tid + i * 256;
        float t = 0.f;
        if (c < Cc) {
            // faithful bug: [B,H,L,D] viewed flat as [B,L,H*D]
            unsigned f = (unsigned)l * Cc + (unsigned)c;
            unsigned hh = f / (Ll * Dd);
            unsigned rem = f - hh * (Ll * Dd);
            unsigned l2 = rem / Dd;
            unsigned d  = rem - l2 * Dd;
            float av = attn[((size_t)(b * Hh + hh) * Ll + l2) * Dd + d];
            t = x[(size_t)row * Cc + c] + av * cb[c];   // alpha1
            x1[(size_t)row * Cc + c] = t;
        }
        v[i] = t; s += t; s2 += t * t;
    }
    red[tid] = s; __syncthreads();
    for (int st = 128; st > 0; st >>= 1) { if (tid < st) red[tid] += red[tid + st]; __syncthreads(); }
    float mu = red[0] * (1.f / Cc); __syncthreads();
    red[tid] = s2; __syncthreads();
    for (int st = 128; st > 0; st >>= 1) { if (tid < st) red[tid] += red[tid + st]; __syncthreads(); }
    float var = red[0] * (1.f / Cc) - mu * mu;
    float rs = rsqrtf(var + 1e-6f);
    #pragma unroll
    for (int i = 0; i < 5; ++i) {
        int c = tid + i * 256;
        if (c < Cc) {
            float y = (v[i] - mu) * rs * lnw[c] + lnb[c];
            y = y * (1.f + cb[5 * Cc + c]) + cb[3 * Cc + c];   // gamma2, beta2
            h2[(size_t)row * Cc + c] = f2bf(y);
        }
    }
}

// ---------------------------------------------------------------- WMMA GEMM: C = A(bf16)[M,K] x BT(bf16)[N,K]^T
// mode 0: +bias, scatter bf16 into qkv [B,H,L,Dp]        (Q,K)
// mode 1: +bias, SiLU, bf16 [M,N]                        (MLP up)
// mode 2: +bias, * alpha2 + x1, fp32 [M,N]               (MLP down)
// mode 3: +bias, scatter bf16 into V^T [B,H,Dp,L]        (V)
__global__ __launch_bounds__(256)
void gemm_bf16_kernel(const u16* __restrict__ A, const u16* __restrict__ BT,
                      const float* __restrict__ bias, int M, int N, int K,
                      int mode, void* __restrict__ outp,
                      const float* __restrict__ cond, const float* __restrict__ x1) {
    __shared__ __align__(16) u16 As[2][128 * 32];
    __shared__ __align__(16) u16 Bs[2][128 * 32];   // [n][k]
    const int tid = threadIdx.x;
    const int lane = tid & 31, lan15 = lane & 15, hi = lane >> 4;
    const int w = tid >> 5, wm = w & 1, wn = w >> 1;   // waves: 2 (M) x 4 (N)
    const int rowBase = blockIdx.y * 128;
    const int colBase = blockIdx.x * 128;

    // staging coords: each thread owns 32 contiguous bytes of A and of B per tile
    const int ldRow = tid >> 1;
    const int ldCol = (tid & 1) * 16;
    const u16* Ag = A  + (size_t)(rowBase + ldRow) * K + ldCol;
    const u16* Bg = BT + (size_t)(colBase + ldRow) * K + ldCol;
    const int ldsOff = ldRow * 32 + ldCol;

    Acc acc[4][2];
    #pragma unroll
    for (int i = 0; i < 4; ++i)
        #pragma unroll
        for (int j = 0; j < 2; ++j)
            #pragma unroll
            for (int r = 0; r < 8; ++r) acc[i][j].f[r] = 0.f;

    const int T = K / 32;
    uint4 ra0, ra1, rb0, rb1;
    ra0 = *(const uint4*)(Ag + 0);  ra1 = *(const uint4*)(Ag + 8);
    rb0 = *(const uint4*)(Bg + 0);  rb1 = *(const uint4*)(Bg + 8);
    *(uint4*)&As[0][ldsOff] = ra0;  *(uint4*)&As[0][ldsOff + 8] = ra1;
    *(uint4*)&Bs[0][ldsOff] = rb0;  *(uint4*)&Bs[0][ldsOff + 8] = rb1;
    __syncthreads();

    for (int t = 0; t < T; ++t) {
        const int cur = t & 1;
        const bool more = (t + 1 < T);
        if (more) {   // prefetch next tile into registers while WMMAs run
            const u16* ap = Ag + (t + 1) * 32;
            const u16* bp = Bg + (t + 1) * 32;
            ra0 = *(const uint4*)(ap + 0);  ra1 = *(const uint4*)(ap + 8);
            rb0 = *(const uint4*)(bp + 0);  rb1 = *(const uint4*)(bp + 8);
        }
        const u16* Asr = &As[cur][0];
        const u16* Bsr = &Bs[cur][0];
        Frag af[4], bfr[2];
        #pragma unroll
        for (int i = 0; i < 4; ++i) {
            int r = wm * 64 + i * 16 + lan15;
            af[i].q[0] = *(const uint4*)&Asr[r * 32 + hi * 8];
            af[i].q[1] = *(const uint4*)&Asr[r * 32 + 16 + hi * 8];
        }
        #pragma unroll
        for (int j = 0; j < 2; ++j) {
            int n = wn * 32 + j * 16 + lan15;
            bfr[j].q[0] = *(const uint4*)&Bsr[n * 32 + hi * 8];
            bfr[j].q[1] = *(const uint4*)&Bsr[n * 32 + 16 + hi * 8];
        }
        #pragma unroll
        for (int i = 0; i < 4; ++i)
            #pragma unroll
            for (int j = 0; j < 2; ++j)
                acc[i][j].v = wmma_bf16(af[i].v, bfr[j].v, acc[i][j].v);
        if (more) {
            const int nb = 1 - cur;
            *(uint4*)&As[nb][ldsOff] = ra0;  *(uint4*)&As[nb][ldsOff + 8] = ra1;
            *(uint4*)&Bs[nb][ldsOff] = rb0;  *(uint4*)&Bs[nb][ldsOff + 8] = rb1;
        }
        __syncthreads();
    }

    #pragma unroll
    for (int i = 0; i < 4; ++i) {
        #pragma unroll
        for (int j = 0; j < 2; ++j) {
            #pragma unroll
            for (int r = 0; r < 8; ++r) {
                int row = rowBase + wm * 64 + i * 16 + r + 8 * hi;
                int col = colBase + wn * 32 + j * 16 + lan15;
                float cv = acc[i][j].f[r] + bias[col];
                if (mode == 0) {
                    int b = row >> 10, l = row & 1023;
                    int hh = col / Dd, d = col - hh * Dd;
                    ((u16*)outp)[((size_t)((b * Hh + hh) * Ll + l)) * Dp + d] = f2bf(cv);
                } else if (mode == 1) {
                    cv = cv / (1.f + __expf(-cv));
                    ((u16*)outp)[(size_t)row * N + col] = f2bf(cv);
                } else if (mode == 2) {
                    int b = row >> 10;
                    ((float*)outp)[(size_t)row * Cc + col] =
                        x1[(size_t)row * Cc + col] + cv * cond[b * C6 + Cc + col]; // alpha2
                } else {   // mode 3: V stored transposed [B,H,Dp,L]
                    int b = row >> 10, l = row & 1023;
                    int hh = col / Dd, d = col - hh * Dd;
                    ((u16*)outp)[((size_t)(b * Hh + hh) * Dp + d) * Ll + l] = f2bf(cv);
                }
            }
        }
    }
}

// ---------------------------------------------------------------- flash attention (bf16 WMMA, f32 accum)
// Q,K: [B,H,L,Dp] bf16 ; V: [B,H,Dp,L] bf16 (pre-transposed) ; O: [B,H,L,Dd] fp32
__global__ __launch_bounds__(256)
void attn_kernel(const u16* __restrict__ Q, const u16* __restrict__ Kb,
                 const u16* __restrict__ Vg, float* __restrict__ O) {
    __shared__ __align__(16) u16 Kt[2][32 * Dp];       // [key][d]
    __shared__ __align__(16) u16 Vt[2][Dp * 32];       // [d][key]
    __shared__ __align__(16) u16 Pst[8 * 16 * 32];     // per-wave P staging
    const int tid = threadIdx.x, lane = tid & 31, lan15 = lane & 15, hi = lane >> 4;
    const int w = tid >> 5;
    const int b = blockIdx.z, h = blockIdx.y, bh = b * Hh + h;
    const int qrow0 = blockIdx.x * 128 + w * 16;

    // staging coords: 384 16B-chunks for K and V each; thread owns chunk tid (+256 if tid<128)
    const bool has1 = (tid < 128);
    const int i1 = tid + 256;
    const int k0key = tid / 12, k0dg = tid - k0key * 12;
    const int k1key = i1 / 12,  k1dg = i1 - k1key * 12;
    const int v0d = tid >> 2, v0g = (tid & 3) * 8;
    const int v1d = i1 >> 2,  v1g = (i1 & 3) * 8;
    const u16* Kgb = Kb + (size_t)bh * Ll * Dp;
    const u16* Vgb = Vg + (size_t)bh * Dp * Ll;

    Frag qf[3];
    {
        const u16* qp = Q + ((size_t)bh * Ll + qrow0 + lan15) * Dp;
        #pragma unroll
        for (int c = 0; c < 3; ++c) {
            qf[c].q[0] = *(const uint4*)(qp + c * 32 + hi * 8);
            qf[c].q[1] = *(const uint4*)(qp + c * 32 + 16 + hi * 8);
        }
    }
    Acc o[6];
    #pragma unroll
    for (int j = 0; j < 6; ++j)
        #pragma unroll
        for (int r = 0; r < 8; ++r) o[j].f[r] = 0.f;
    float mrow[8], rsum[8];
    #pragma unroll
    for (int r = 0; r < 8; ++r) { mrow[r] = -1e30f; rsum[r] = 0.f; }
    const float scale = 0.117851130f;   // 1/sqrt(72)

    uint4 rk0, rk1, rv0, rv1;
    rk0 = *(const uint4*)(Kgb + (size_t)k0key * Dp + k0dg * 8);
    rv0 = *(const uint4*)(Vgb + (size_t)v0d * Ll + v0g);
    if (has1) {
        rk1 = *(const uint4*)(Kgb + (size_t)k1key * Dp + k1dg * 8);
        rv1 = *(const uint4*)(Vgb + (size_t)v1d * Ll + v1g);
    }
    *(uint4*)&Kt[0][k0key * Dp + k0dg * 8] = rk0;
    *(uint4*)&Vt[0][v0d * 32 + v0g] = rv0;
    if (has1) {
        *(uint4*)&Kt[0][k1key * Dp + k1dg * 8] = rk1;
        *(uint4*)&Vt[0][v1d * 32 + v1g] = rv1;
    }
    __syncthreads();

    for (int kt = 0; kt < Ll / 32; ++kt) {
        const int cur = kt & 1;
        const bool more = (kt + 1 < Ll / 32);
        if (more) {
            const u16* kg = Kgb + (size_t)(kt + 1) * 32 * Dp;
            const u16* vg = Vgb + (kt + 1) * 32;
            rk0 = *(const uint4*)(kg + (size_t)k0key * Dp + k0dg * 8);
            rv0 = *(const uint4*)(vg + (size_t)v0d * Ll + v0g);
            if (has1) {
                rk1 = *(const uint4*)(kg + (size_t)k1key * Dp + k1dg * 8);
                rv1 = *(const uint4*)(vg + (size_t)v1d * Ll + v1g);
            }
        }

        Acc s0, s1;
        #pragma unroll
        for (int r = 0; r < 8; ++r) { s0.f[r] = 0.f; s1.f[r] = 0.f; }
        const u16* Kcur = &Kt[cur][0];
        const u16* Vcur = &Vt[cur][0];
        #pragma unroll
        for (int c = 0; c < 3; ++c) {
            Frag bk;
            const u16* kp0 = &Kcur[lan15 * Dp + c * 32 + hi * 8];
            bk.q[0] = *(const uint4*)kp0; bk.q[1] = *(const uint4*)(kp0 + 16);
            s0.v = wmma_bf16(qf[c].v, bk.v, s0.v);
            const u16* kp1 = &Kcur[(16 + lan15) * Dp + c * 32 + hi * 8];
            bk.q[0] = *(const uint4*)kp1; bk.q[1] = *(const uint4*)(kp1 + 16);
            s1.v = wmma_bf16(qf[c].v, bk.v, s1.v);
        }

        u16* pw = &Pst[w * 512];
        #pragma unroll
        for (int r = 0; r < 8; ++r) {
            float a0 = s0.f[r] * scale, a1 = s1.f[r] * scale;
            float t = fmaxf(a0, a1);
            #pragma unroll
            for (int m = 1; m < 16; m <<= 1) t = fmaxf(t, __shfl_xor(t, m, 32));
            float newm = fmaxf(mrow[r], t);
            float f = __expf(mrow[r] - newm);
            mrow[r] = newm;
            float p0 = __expf(a0 - newm), p1 = __expf(a1 - newm);
            float ps = p0 + p1;
            #pragma unroll
            for (int m = 1; m < 16; m <<= 1) ps += __shfl_xor(ps, m, 32);
            rsum[r] = rsum[r] * f + ps;
            #pragma unroll
            for (int j = 0; j < 6; ++j) o[j].f[r] *= f;
            int prow = r + 8 * hi;
            pw[prow * 32 + lan15]      = f2bf(p0);
            pw[prow * 32 + 16 + lan15] = f2bf(p1);
        }
        Frag pf;   // C-layout -> A-layout via same-wave LDS roundtrip
        pf.q[0] = *(const uint4*)&pw[lan15 * 32 + hi * 8];
        pf.q[1] = *(const uint4*)&pw[lan15 * 32 + 16 + hi * 8];
        #pragma unroll
        for (int j = 0; j < 6; ++j) {
            Frag bv;
            const u16* vp = &Vcur[(j * 16 + lan15) * 32 + hi * 8];
            bv.q[0] = *(const uint4*)vp; bv.q[1] = *(const uint4*)(vp + 16);
            o[j].v = wmma_bf16(pf.v, bv.v, o[j].v);
        }

        if (more) {
            const int nb = 1 - cur;
            *(uint4*)&Kt[nb][k0key * Dp + k0dg * 8] = rk0;
            *(uint4*)&Vt[nb][v0d * 32 + v0g] = rv0;
            if (has1) {
                *(uint4*)&Kt[nb][k1key * Dp + k1dg * 8] = rk1;
                *(uint4*)&Vt[nb][v1d * 32 + v1g] = rv1;
            }
        }
        __syncthreads();
    }

    #pragma unroll
    for (int r = 0; r < 8; ++r) {
        float inv = 1.f / rsum[r];
        int row = qrow0 + r + 8 * hi;
        #pragma unroll
        for (int j = 0; j < 5; ++j) {
            int col = j * 16 + lan15;
            if (col < Dd)
                O[((size_t)bh * Ll + row) * Dd + col] = o[j].f[r] * inv;
        }
    }
}

// ----------------------------------------------------------------
extern "C" void kernel_launch(void* const* d_in, const int* in_sizes, int n_in,
                              void* d_out, int out_size, void* d_ws, size_t ws_size,
                              hipStream_t stream) {
    (void)in_sizes; (void)n_in; (void)out_size; (void)ws_size;
    const float* x         = (const float*)d_in[0];
    const float* condition = (const float*)d_in[1];
    const float* w_q = (const float*)d_in[2];  const float* b_q = (const float*)d_in[3];
    const float* w_k = (const float*)d_in[4];  const float* b_k = (const float*)d_in[5];
    const float* w_v = (const float*)d_in[6];  const float* b_v = (const float*)d_in[7];
    const float* ln1w = (const float*)d_in[8];  const float* ln1b = (const float*)d_in[9];
    const float* ln2w = (const float*)d_in[10]; const float* ln2b = (const float*)d_in[11];
    const float* w1 = (const float*)d_in[12]; const float* b1 = (const float*)d_in[13];
    const float* w2 = (const float*)d_in[14]; const float* b2 = (const float*)d_in[15];
    const float* adaw = (const float*)d_in[16]; const float* adab = (const float*)d_in[17];

    char* ws = (char*)d_ws;
    size_t off = 0;
    auto alloc = [&](size_t bytes) -> void* {
        void* p = ws + off;
        off += (bytes + 255) & ~(size_t)255;
        return p;
    };
    float* cond = (float*)alloc((size_t)Bb * C6 * 4);
    u16* wqT = (u16*)alloc((size_t)Cc * Cc * 2);   // [N][K] bf16
    u16* wkT = (u16*)alloc((size_t)Cc * Cc * 2);
    u16* wvT = (u16*)alloc((size_t)Cc * Cc * 2);
    u16* w1T = (u16*)alloc((size_t)C4 * Cc * 2);   // [4C][C]
    u16* w2T = (u16*)alloc((size_t)Cc * C4 * 2);   // [C][4C]
    u16* h_bf  = (u16*)alloc((size_t)Mr * Cc * 2); // reused as h2 after attention
    u16* q_bf  = (u16*)alloc((size_t)Bb * Hh * Ll * Dp * 2);
    u16* k_bf  = (u16*)alloc((size_t)Bb * Hh * Ll * Dp * 2);
    u16* v_bf  = (u16*)alloc((size_t)Bb * Hh * Dp * Ll * 2);   // transposed
    float* attn = (float*)alloc((size_t)Bb * Hh * Ll * Dd * 4);
    float* x1   = (float*)alloc((size_t)Mr * Cc * 4);
    u16* m_bf   = (u16*)alloc((size_t)Mr * C4 * 2);

    // 1. weights -> bf16, pre-transposed to [N][K]
    transpose_f2bf_kernel<<<dim3(Cc / 32, Cc / 32), 256, 0, stream>>>(w_q, wqT, Cc, Cc);
    transpose_f2bf_kernel<<<dim3(Cc / 32, Cc / 32), 256, 0, stream>>>(w_k, wkT, Cc, Cc);
    transpose_f2bf_kernel<<<dim3(Cc / 32, Cc / 32), 256, 0, stream>>>(w_v, wvT, Cc, Cc);
    transpose_f2bf_kernel<<<dim3(Cc / 32, C4 / 32), 256, 0, stream>>>(w1, w1T, Cc, C4);
    transpose_f2bf_kernel<<<dim3(C4 / 32, Cc / 32), 256, 0, stream>>>(w2, w2T, C4, Cc);

    // 2. adaLN conditioning
    ada_kernel<<<C6 / 256, 256, 0, stream>>>(condition, adaw, adab, cond);

    // 3. LN1 + modulate
    ln1_mod_kernel<<<Mr, 256, 0, stream>>>(x, cond, ln1w, ln1b, h_bf);

    // 4. QKV projections (pad regions zeroed for the K-padded WMMA loop)
    size_t qkvBytes = (size_t)Bb * Hh * Ll * Dp * 2;
    hipMemsetAsync(q_bf, 0, qkvBytes, stream);
    hipMemsetAsync(k_bf, 0, qkvBytes, stream);
    hipMemsetAsync(v_bf, 0, qkvBytes, stream);
    dim3 gqkv(Cc / 128, Mr / 128);
    gemm_bf16_kernel<<<gqkv, 256, 0, stream>>>(h_bf, wqT, b_q, Mr, Cc, Cc, 0, q_bf, nullptr, nullptr);
    gemm_bf16_kernel<<<gqkv, 256, 0, stream>>>(h_bf, wkT, b_k, Mr, Cc, Cc, 0, k_bf, nullptr, nullptr);
    gemm_bf16_kernel<<<gqkv, 256, 0, stream>>>(h_bf, wvT, b_v, Mr, Cc, Cc, 3, v_bf, nullptr, nullptr);

    // 5. attention
    attn_kernel<<<dim3(Ll / 128, Hh, Bb), 256, 0, stream>>>(q_bf, k_bf, v_bf, attn);

    // 6. residual (faithful reshape bug) + LN2 + modulate  (h_bf reused as h2)
    resid_ln2_kernel<<<Mr, 256, 0, stream>>>(x, attn, cond, ln2w, ln2b, x1, h_bf);

    // 7. MLP up + SiLU
    gemm_bf16_kernel<<<dim3(C4 / 128, Mr / 128), 256, 0, stream>>>(
        h_bf, w1T, b1, Mr, C4, Cc, 1, m_bf, nullptr, nullptr);

    // 8. MLP down + alpha2 residual -> d_out (fp32)
    gemm_bf16_kernel<<<dim3(Cc / 128, Mr / 128), 256, 0, stream>>>(
        m_bf, w2T, b2, Mr, Cc, C4, 2, d_out, cond, x1);
}
